// Sequence_48180943127334
// MI455X (gfx1250) — matrix operand order repaired and checked by reading, not compiled, not run on hardware
//
#include <hip/hip_runtime.h>

typedef __attribute__((ext_vector_type(16))) _Float16 v16h;
typedef __attribute__((ext_vector_type(8)))  _Float16 v8h;
typedef __attribute__((ext_vector_type(8)))  float    v8f;

#define HID   100
#define GP    112            // per-gate padded width (7 tiles of 16)
#define NG    448            // 4 gates * GP
#define KROW  104            // padded K elements per weight row
#define KROWB 208            // bytes per weight row (104 * 2)
#define HROWB 256            // bytes per h row (128 f16, cols >=100 are zero)

// ---- shared memory layout (bytes) ----
#define OFF_W1    0
#define WSZ       (NG * KROWB)            // 93184 per matrix
#define OFF_W2I   (WSZ)
#define OFF_W2H   (2 * WSZ)
#define OFF_WIH1  (3 * WSZ)               // f16[448]
#define OFF_WLIN  (OFF_WIH1 + 896)        // f16[112]
#define OFF_B1    (OFF_WLIN + 224)        // f32[448]
#define OFF_B2    (OFF_B1 + 1792)         // f32[448]
#define OFF_WAVE  (OFF_B2 + 1792)         // 284256
#define WAVE_SZ   8256                    // h1 4096 + h2 4096 + xbuf 64
#define SMEM_TOTAL (OFF_WAVE + 4 * WAVE_SZ)   // 317280 <= 320KB

__device__ __forceinline__ float sigm(float x) {
    return 1.0f / (1.0f + __expf(-x));
}
__device__ __forceinline__ float tanh_(float x) {
    float e = __expf(2.0f * x);
    return (e - 1.0f) / (e + 1.0f);
}

// A fragment (16x32 f16): lane lm = row M; K = 8*half+[0..7], 16+8*half+[0..7].
__device__ __forceinline__ v16h load_frag_A(const char* base, int lm, int half, int kc) {
    const char* p = base + lm * HROWB + kc * 64 + half * 16;
    v8h lo = *(const v8h*)p;
    v8h hi = *(const v8h*)(p + 32);
    return __builtin_shufflevector(lo, hi, 0,1,2,3,4,5,6,7,8,9,10,11,12,13,14,15);
}
// B fragment (32x16 f16) from [N][K] rows: lane lm = col N; K = 8*half+[0..7], 16+8*half+[0..7].
__device__ __forceinline__ v16h load_frag_B(const char* wbase, int nrow, int half, int kc) {
    const char* p = wbase + nrow * KROWB + kc * 64 + half * 16;
    v8h lo = *(const v8h*)p;
    v8h hi = *(const v8h*)(p + 32);
    return __builtin_shufflevector(lo, hi, 0,1,2,3,4,5,6,7,8,9,10,11,12,13,14,15);
}

__global__ __launch_bounds__(128, 1)
void lstm_persistent_kernel(
    const float* __restrict__ input,   // [B, T]
    const float* __restrict__ Wih1,    // [400,1]
    const float* __restrict__ Whh1,    // [400,100]
    const float* __restrict__ bih1,
    const float* __restrict__ bhh1,
    const float* __restrict__ Wih2,    // [400,100]
    const float* __restrict__ Whh2,    // [400,100]
    const float* __restrict__ bih2,
    const float* __restrict__ bhh2,
    const float* __restrict__ Wlin,    // [1,100]
    const float* __restrict__ blin,    // [1]
    const int*   __restrict__ futp,    // [1]
    float*       __restrict__ out,     // [B, T+future]
    int T)
{
    extern __shared__ char smem[];
    const int tid = threadIdx.x;

    // ---- one-time: zero LDS (pads must be 0), then stage f16 weights ----
    for (int i = tid; i < SMEM_TOTAL / 4; i += 128)
        ((int*)smem)[i] = 0;
    __syncthreads();

    {
        _Float16* w1  = (_Float16*)(smem + OFF_W1);
        _Float16* w2i = (_Float16*)(smem + OFF_W2I);
        _Float16* w2h = (_Float16*)(smem + OFF_W2H);
        for (int i = tid; i < 400 * HID; i += 128) {
            int j = i / HID, k = i - j * HID;
            int r = (j / 100) * GP + (j % 100);
            w1 [r * KROW + k] = (_Float16)Whh1[i];
            w2i[r * KROW + k] = (_Float16)Wih2[i];
            w2h[r * KROW + k] = (_Float16)Whh2[i];
        }
        _Float16* wih1 = (_Float16*)(smem + OFF_WIH1);
        float* b1 = (float*)(smem + OFF_B1);
        float* b2 = (float*)(smem + OFF_B2);
        for (int i = tid; i < 400; i += 128) {
            int r = (i / 100) * GP + (i % 100);
            wih1[r] = (_Float16)Wih1[i];
            b1[r] = bih1[i] + bhh1[i];
            b2[r] = bih2[i] + bhh2[i];
        }
        _Float16* wlin = (_Float16*)(smem + OFF_WLIN);
        for (int i = tid; i < HID; i += 128)
            wlin[i] = (_Float16)Wlin[i];
    }
    __syncthreads();
    // ---- after this point: each wave is fully independent, no barriers ----

    const int wv   = tid >> 5;
    const int lane = tid & 31;
    const int half = lane >> 4;
    const int lm   = lane & 15;

    char*  wbase = smem + OFF_WAVE + wv * WAVE_SZ;
    char*  H1 = wbase;
    char*  H2 = wbase + 4096;
    float* XB = (float*)(wbase + 8192);

    const char* W1p  = smem + OFF_W1;
    const char* W2Ip = smem + OFF_W2I;
    const char* W2Hp = smem + OFF_W2H;
    const float* B1p = (const float*)(smem + OFF_B1);
    const float* B2p = (const float*)(smem + OFF_B2);
    const _Float16* WIH1p = (const _Float16*)(smem + OFF_WIH1);
    const _Float16* WLINp = (const _Float16*)(smem + OFF_WLIN);

    const int row0 = (blockIdx.x * 4 + wv) * 16;
    const int fut  = futp[0];
    const int Ttot = T + fut;
    const float blinv = blin[0];

    // cell state packed as f16 fragments: 28 VGPRs per layer (vs 56 as f32)
    v8h c1[7], c2[7];
#pragma unroll
    for (int t6 = 0; t6 < 7; ++t6)
#pragma unroll
        for (int g = 0; g < 8; ++g) { c1[t6][g] = (_Float16)0.0f; c2[t6][g] = (_Float16)0.0f; }

    for (int t = 0; t < Ttot; ++t) {
        // stage x: main phase from input; future phase XB already holds y
        if (t < T) {
            if (lane < 16) XB[lane] = input[(row0 + lane) * T + t];
        }

        // ---------------- layer 1 ----------------
        v16h a1[4];
#pragma unroll
        for (int kc = 0; kc < 4; ++kc) a1[kc] = load_frag_A(H1, lm, half, kc);
        float xg[8];
#pragma unroll
        for (int g = 0; g < 8; ++g) xg[g] = XB[half * 8 + g];

#pragma unroll
        for (int t6 = 0; t6 < 7; ++t6) {
            v8f acc[4];
#pragma unroll
            for (int gi = 0; gi < 4; ++gi) {
                const int N = gi * GP + t6 * 16 + lm;
                // preload all 4 B fragments before the WMMA chain (hide LDS latency)
                v16h bfr[4];
#pragma unroll
                for (int kc = 0; kc < 4; ++kc) bfr[kc] = load_frag_B(W1p, N, half, kc);
                const float bv = B1p[N];
                const float wiv = (float)WIH1p[N];
                v8f a;
#pragma unroll
                for (int g = 0; g < 8; ++g) a[g] = bv + xg[g] * wiv;
#pragma unroll
                for (int kc = 0; kc < 4; ++kc) {
                    a = __builtin_amdgcn_wmma_f32_16x16x32_f16(
                            false, a1[kc], false, bfr[kc], (short)0, a, false, false);
                }
                acc[gi] = a;
            }
            const int col = t6 * 16 + lm;
            const bool valid = col < HID;
#pragma unroll
            for (int g = 0; g < 8; ++g) {
                float ig = sigm(acc[0][g]);
                float fg = sigm(acc[1][g]);
                float gg = tanh_(acc[2][g]);
                float og = sigm(acc[3][g]);
                float c = fg * (float)c1[t6][g] + ig * gg;
                c1[t6][g] = (_Float16)c;
                float h = og * tanh_(c);
                h = valid ? h : 0.0f;
                ((_Float16*)(H1 + (g + half * 8) * HROWB))[col] = (_Float16)h;
            }
        }

        // ---------------- layer 2 ----------------
        v16h ah1[4], ah2[4];
#pragma unroll
        for (int kc = 0; kc < 4; ++kc) {
            ah1[kc] = load_frag_A(H1, lm, half, kc);   // new h1
            ah2[kc] = load_frag_A(H2, lm, half, kc);   // previous h2
        }
#pragma unroll
        for (int t6 = 0; t6 < 7; ++t6) {
            v8f acc[4];
#pragma unroll
            for (int gi = 0; gi < 4; ++gi) {
                const int N = gi * GP + t6 * 16 + lm;
                const float bv = B2p[N];
                v8f a;
#pragma unroll
                for (int g = 0; g < 8; ++g) a[g] = bv;
                // preload W_ih2 fragments, run its chain; then W_hh2 likewise
                {
                    v16h bfr[4];
#pragma unroll
                    for (int kc = 0; kc < 4; ++kc) bfr[kc] = load_frag_B(W2Ip, N, half, kc);
#pragma unroll
                    for (int kc = 0; kc < 4; ++kc) {
                        a = __builtin_amdgcn_wmma_f32_16x16x32_f16(
                                false, ah1[kc], false, bfr[kc], (short)0, a, false, false);
                    }
                }
                {
                    v16h bfr[4];
#pragma unroll
                    for (int kc = 0; kc < 4; ++kc) bfr[kc] = load_frag_B(W2Hp, N, half, kc);
#pragma unroll
                    for (int kc = 0; kc < 4; ++kc) {
                        a = __builtin_amdgcn_wmma_f32_16x16x32_f16(
                                false, ah2[kc], false, bfr[kc], (short)0, a, false, false);
                    }
                }
                acc[gi] = a;
            }
            const int col = t6 * 16 + lm;
            const bool valid = col < HID;
#pragma unroll
            for (int g = 0; g < 8; ++g) {
                float ig = sigm(acc[0][g]);
                float fg = sigm(acc[1][g]);
                float gg = tanh_(acc[2][g]);
                float og = sigm(acc[3][g]);
                float c = fg * (float)c2[t6][g] + ig * gg;
                c2[t6][g] = (_Float16)c;
                float h = og * tanh_(c);
                h = valid ? h : 0.0f;
                ((_Float16*)(H2 + (g + half * 8) * HROWB))[col] = (_Float16)h;
            }
        }

        // ---------------- output projection ----------------
        float s = 0.0f;
        const _Float16* h2r = (const _Float16*)(H2 + lm * HROWB);
#pragma unroll
        for (int j = 0; j < 7; ++j) {
            v8h hv = *(const v8h*)(h2r + half * 56 + j * 8);
            v8h wl = *(const v8h*)(WLINp + half * 56 + j * 8);
#pragma unroll
            for (int i = 0; i < 8; ++i) s += (float)hv[i] * (float)wl[i];
        }
        s += __shfl_xor(s, 16, 32);
        const float y = s + blinv;
        if (lane < 16) {
            out[(row0 + lane) * Ttot + t] = y;
            XB[lane] = y;                    // feeds autoregressive phase
        }
    }
}

extern "C" void kernel_launch(void* const* d_in, const int* in_sizes, int n_in,
                              void* d_out, int out_size, void* d_ws, size_t ws_size,
                              hipStream_t stream) {
    const float* input = (const float*)d_in[0];
    const float* Wih1  = (const float*)d_in[1];
    const float* Whh1  = (const float*)d_in[2];
    const float* bih1  = (const float*)d_in[3];
    const float* bhh1  = (const float*)d_in[4];
    const float* Wih2  = (const float*)d_in[5];
    const float* Whh2  = (const float*)d_in[6];
    const float* bih2  = (const float*)d_in[7];
    const float* bhh2  = (const float*)d_in[8];
    const float* Wlin  = (const float*)d_in[9];
    const float* blin  = (const float*)d_in[10];
    const int*   futp  = (const int*)d_in[11];

    const int B = 512;
    const int T = in_sizes[0] / B;

    dim3 grid(B / 64);      // 8 workgroups, 4 waves each, 16 batch rows per wave
    dim3 block(128);
    lstm_persistent_kernel<<<grid, block, SMEM_TOTAL, stream>>>(
        input, Wih1, Whh1, bih1, bhh1, Wih2, Whh2, bih2, bhh2,
        Wlin, blin, futp, (float*)d_out, T);
}
